// GINClassifier_88888643158683
// MI455X (gfx1250) — matrix operand order repaired and checked by reading, not compiled, not run on hardware
//
#include <hip/hip_runtime.h>

#define NNODES 50000
#define NEDGES 800000
#define NGRAPHS 128
#define D 64
#define DOUT 16
#define NLAYERS 4
#define BN_EPS 1e-5f

typedef __bf16 bf16;
typedef bf16  v16bf __attribute__((ext_vector_type(16)));
typedef bf16  v8bf  __attribute__((ext_vector_type(8)));
typedef float v8f   __attribute__((ext_vector_type(8)));
typedef float v8fa  __attribute__((ext_vector_type(8)));

// ---------------- utility kernels ----------------
__global__ void zero_f32(float* __restrict__ p, long n) {
    long i = (long)blockIdx.x * blockDim.x + threadIdx.x;
    if (i < n) p[i] = 0.0f;
}

__global__ void copy_f32(float* __restrict__ dst, const float* __restrict__ src, long n) {
    long i = (long)blockIdx.x * blockDim.x + threadIdx.x;
    if (i < n) dst[i] = src[i];
}

// ---------------- pre-swizzle weights into WMMA B-fragment order ----------------
// Fragment layout: Wf[((m*ntiles + nt)*2 + kc)*32*16 + lane*16 + j]
//   holds W[m][k][n] with n = nt*16 + (lane&15), k = kc*32 + (lane>>4)*16 + j
// so each lane's 16-element B fragment is one contiguous 32-byte load.
__global__ void swizzle_wb(const float* __restrict__ W, bf16* __restrict__ Wf,
                           int N, int nmat) {
    int ntiles = N / 16;
    long total = (long)nmat * ntiles * 2 * 32 * 16;
    long idx = (long)blockIdx.x * blockDim.x + threadIdx.x;
    if (idx >= total) return;
    int j    = (int)(idx & 15);
    int lane = (int)((idx >> 4) & 31);
    int kc   = (int)((idx >> 9) & 1);
    long rest = idx >> 10;
    int nt = (int)(rest % ntiles);
    int m  = (int)(rest / ntiles);
    int n = nt * 16 + (lane & 15);
    int k = kc * 32 + (lane >> 4) * 16 + j;
    Wf[idx] = (bf16)W[((long)m * D + k) * N + n];
}

// ---------------- edge scatter: agg[dst] += w[e] * h[src] ----------------
// One thread per (edge, float4). Working set fits in 192MB L2 -> atomics hit L2.
__global__ void scatter_edges(const float* __restrict__ h, const float* __restrict__ w,
                              const int* __restrict__ src, const int* __restrict__ dst,
                              float* __restrict__ agg) {
    long idx = (long)blockIdx.x * blockDim.x + threadIdx.x;
    if (idx >= (long)NEDGES * (D / 4)) return;
    int e  = (int)(idx >> 4);
    int f4 = ((int)idx & 15) << 2;
    int s = src[e];
    int d = dst[e];
    float wv = w[e];
    const float4 hv = *(const float4*)(h + (long)s * D + f4);
    float* ap = agg + (long)d * D + f4;
    atomicAdd(ap + 0, wv * hv.x);
    atomicAdd(ap + 1, wv * hv.y);
    atomicAdd(ap + 2, wv * hv.z);
    atomicAdd(ap + 3, wv * hv.w);
}

// ---------------- h = (1+eps[l])*h + agg ----------------
__global__ void gin_combine(float* __restrict__ h, const float* __restrict__ agg,
                            const float* __restrict__ eps, int l, long n) {
    long i = (long)blockIdx.x * blockDim.x + threadIdx.x;
    if (i >= n) return;
    float e1 = 1.0f + eps[l];
    h[i] = e1 * h[i] + agg[i];
}

// ---------------- BatchNorm column stats: stats[0:64]=sum, [64:128]=sumsq --------
__global__ void bn_stats(const float* __restrict__ X, float* __restrict__ stats, int nrows) {
    int f  = threadIdx.x;               // 64 threads, one per feature column
    int r0 = blockIdx.x * 512;
    int r1 = r0 + 512; if (r1 > nrows) r1 = nrows;
    float s = 0.0f, q = 0.0f;
    for (int r = r0; r < r1; ++r) {
        float v = X[(long)r * D + f];
        s += v; q += v * v;
    }
    atomicAdd(&stats[f], s);
    atomicAdd(&stats[D + f], q);
}

// ---------------- y = relu(gamma*(x-mu)*rsqrt(var+eps)+beta) ----------------
__global__ void bn_apply_relu(const float* __restrict__ X, float* __restrict__ Y,
                              const float* __restrict__ stats,
                              const float* __restrict__ gamma, const float* __restrict__ beta,
                              int nrows) {
    long i = (long)blockIdx.x * blockDim.x + threadIdx.x;
    if (i >= (long)nrows * D) return;
    int f = (int)(i & (D - 1));
    float inv_n = 1.0f / (float)nrows;
    float mu  = stats[f] * inv_n;
    float var = stats[D + f] * inv_n - mu * mu;
    float y = gamma[f] * (X[i] - mu) * rsqrtf(var + BN_EPS) + beta[f];
    Y[i] = y > 0.0f ? y : 0.0f;
}

// ---------------- graph pooling (graph_ids sorted): pooled[g] += sum of rows ----
__global__ void pool_sorted(const float* __restrict__ H, const int* __restrict__ gid,
                            float* __restrict__ pooled, int nrows) {
    int f  = threadIdx.x;               // 64 threads
    int r0 = blockIdx.x * 256;
    if (r0 >= nrows) return;
    int r1 = r0 + 256; if (r1 > nrows) r1 = nrows;
    int curg = gid[r0];
    float acc = 0.0f;
    for (int r = r0; r < r1; ++r) {
        int g = gid[r];
        if (g != curg) {
            atomicAdd(&pooled[curg * D + f], acc);
            acc = 0.0f; curg = g;
        }
        acc += H[(long)r * D + f];
    }
    atomicAdd(&pooled[curg * D + f], acc);
}

// ---------------- WMMA GEMM: OUT[M,N] (+)= A[M,64](f32->bf16) @ Wfrag(bf16) + bias ----
// One wave per 16-row M tile; K=64 as two k=32 WMMA steps; sweeps all N tiles.
// B fragments are pre-swizzled (swizzle_wb) -> one contiguous 32B load per fragment.
template <bool ACCUM>
__global__ void wmma_gemm_k64(const float* __restrict__ A, const bf16* __restrict__ Wf,
                              const float* __restrict__ bias, float* __restrict__ OUT,
                              int Mtiles, int N) {
    int lane = threadIdx.x & 31;
    int wid  = threadIdx.x >> 5;
    int tile = blockIdx.x * 4 + wid;
    if (tile >= Mtiles) return;          // wave-uniform exit: live waves keep EXEC all-1
    int m0    = tile * 16;
    int khalf = lane >> 4;               // lane group 0/1
    int lrow  = lane & 15;
    int mrow  = m0 + lrow;

    // A fragments in contiguous k order (matches swizzle_wb B): k = khalf*16 + j.
    // Two 32B vector loads + packed f32->bf16 conversion per fragment.
    v16bf a[2];
#pragma unroll
    for (int kc = 0; kc < 2; ++kc) {
        const float* base = A + (long)mrow * 64 + kc * 32 + khalf * 16;
        v8f f0 = *(const v8f*)(base);        // j = 0..7
        v8f f1 = *(const v8f*)(base + 8);    // j = 8..15
        v8bf c0 = __builtin_convertvector(f0, v8bf);
        v8bf c1 = __builtin_convertvector(f1, v8bf);
        a[kc] = __builtin_shufflevector(c0, c1, 0, 1, 2, 3, 4, 5, 6, 7,
                                                8, 9, 10, 11, 12, 13, 14, 15);
    }

    const v16bf* Bf = (const v16bf*)Wf;      // 32B-aligned fragment array
    for (int nt = 0; nt < N / 16; ++nt) {
        v16bf b0 = Bf[(nt * 2 + 0) * 32 + lane];
        v16bf b1 = Bf[(nt * 2 + 1) * 32 + lane];

        // C/D f32 layout: VGPR v -> row m0 + v + 8*khalf, col n
        int n = nt * 16 + lrow;
        float bv = bias[n];
        v8f c;
#pragma unroll
        for (int v = 0; v < 8; ++v) {
            int m = m0 + v + khalf * 8;
            c[v] = (ACCUM ? OUT[(long)m * N + n] : 0.0f) + bv;
        }
        c = __builtin_amdgcn_wmma_f32_16x16x32_bf16(false, a[0], false, b0, (short)0, c, false, false);
        c = __builtin_amdgcn_wmma_f32_16x16x32_bf16(false, a[1], false, b1, (short)0, c, false, false);
#pragma unroll
        for (int v = 0; v < 8; ++v) {
            int m = m0 + v + khalf * 8;
            OUT[(long)m * N + n] = c[v];
        }
    }
}

static inline int cdiv(long a, long b) { return (int)((a + b - 1) / b); }

extern "C" void kernel_launch(void* const* d_in, const int* in_sizes, int n_in,
                              void* d_out, int out_size, void* d_ws, size_t ws_size,
                              hipStream_t stream) {
    (void)in_sizes; (void)n_in; (void)out_size; (void)ws_size;
    const float* x    = (const float*)d_in[0];
    const float* w    = (const float*)d_in[1];
    const int*   src  = (const int*)d_in[2];
    const int*   dst  = (const int*)d_in[3];
    const int*   gid  = (const int*)d_in[4];
    const float* eps  = (const float*)d_in[5];
    const float* W1   = (const float*)d_in[6];
    const float* b1   = (const float*)d_in[7];
    const float* bng1 = (const float*)d_in[8];
    const float* bnb1 = (const float*)d_in[9];
    const float* W2   = (const float*)d_in[10];
    const float* b2   = (const float*)d_in[11];
    const float* bng2 = (const float*)d_in[12];
    const float* bnb2 = (const float*)d_in[13];
    const float* PW   = (const float*)d_in[14];
    const float* Pb   = (const float*)d_in[15];
    float* score = (float*)d_out;

    const long ND = (long)NNODES * D;
    float* h      = (float*)d_ws;
    float* t      = h + ND;
    float* r      = t + ND;                 // doubles as edge-aggregate buffer
    float* pooled = r + ND;                 // [128,64]
    float* stats  = pooled + NGRAPHS * D;   // [128]
    bf16*  W1f    = (bf16*)(stats + 2 * D); // 32B-aligned (byte offset % 32 == 0)
    bf16*  W2f    = W1f + (long)NLAYERS * D * D;
    bf16*  PWf    = W2f + (long)NLAYERS * D * D;

    // one-time weight swizzle into WMMA fragment order (deterministic per call)
    swizzle_wb<<<cdiv((long)NLAYERS * D * D, 256), 256, 0, stream>>>(W1, W1f, D, NLAYERS);
    swizzle_wb<<<cdiv((long)NLAYERS * D * D, 256), 256, 0, stream>>>(W2, W2f, D, NLAYERS);
    swizzle_wb<<<cdiv((long)(NLAYERS + 1) * D * DOUT, 256), 256, 0, stream>>>(PW, PWf, DOUT, NLAYERS + 1);

    copy_f32<<<cdiv(ND, 256), 256, 0, stream>>>(h, x, ND);

    // layer-0 prediction on raw features
    zero_f32<<<cdiv((long)NGRAPHS * D, 256), 256, 0, stream>>>(pooled, (long)NGRAPHS * D);
    pool_sorted<<<cdiv(NNODES, 256), 64, 0, stream>>>(x, gid, pooled, NNODES);
    wmma_gemm_k64<false><<<cdiv(NGRAPHS / 16, 4), 128, 0, stream>>>(pooled, PWf, Pb, score, NGRAPHS / 16, DOUT);

    const int mtiles = NNODES / 16;  // 3125
    for (int l = 0; l < NLAYERS; ++l) {
        // aggregate neighbors
        zero_f32<<<cdiv(ND, 256), 256, 0, stream>>>(r, ND);
        scatter_edges<<<cdiv((long)NEDGES * 16, 256), 256, 0, stream>>>(h, w, src, dst, r);
        gin_combine<<<cdiv(ND, 256), 256, 0, stream>>>(h, r, eps, l, ND);

        // Linear1 -> BN -> ReLU
        wmma_gemm_k64<false><<<cdiv(mtiles, 4), 128, 0, stream>>>(h, W1f + (long)l * D * D, b1 + l * D, t, mtiles, D);
        zero_f32<<<1, 128, 0, stream>>>(stats, 2 * D);
        bn_stats<<<cdiv(NNODES, 512), 64, 0, stream>>>(t, stats, NNODES);
        bn_apply_relu<<<cdiv(ND, 256), 256, 0, stream>>>(t, r, stats, bng1 + l * D, bnb1 + l * D, NNODES);

        // Linear2 -> outer BN -> ReLU
        wmma_gemm_k64<false><<<cdiv(mtiles, 4), 128, 0, stream>>>(r, W2f + (long)l * D * D, b2 + l * D, t, mtiles, D);
        zero_f32<<<1, 128, 0, stream>>>(stats, 2 * D);
        bn_stats<<<cdiv(NNODES, 512), 64, 0, stream>>>(t, stats, NNODES);
        bn_apply_relu<<<cdiv(ND, 256), 256, 0, stream>>>(t, h, stats, bng2 + l * D, bnb2 + l * D, NNODES);

        // readout + prediction head (accumulate into score)
        zero_f32<<<cdiv((long)NGRAPHS * D, 256), 256, 0, stream>>>(pooled, (long)NGRAPHS * D);
        pool_sorted<<<cdiv(NNODES, 256), 64, 0, stream>>>(h, gid, pooled, NNODES);
        wmma_gemm_k64<true><<<cdiv(NGRAPHS / 16, 4), 128, 0, stream>>>(
            pooled, PWf + (long)(l + 1) * D * DOUT, Pb + (l + 1) * DOUT, score, NGRAPHS / 16, DOUT);
    }
}
